// TemporalLinkPredictionModel_43542378447127
// MI455X (gfx1250) — compile-verified, matrix operand order checked by reading, not run on hardware
//
#include <hip/hip_runtime.h>
#include <hip/hip_bf16.h>

#define NN  1024
#define EE  32768
#define FN  300
#define FNP 320
#define CC  128

typedef __bf16 bf16_t;
typedef __attribute__((ext_vector_type(16))) __bf16 v16bf;
typedef __attribute__((ext_vector_type(8)))  float  v8f;
typedef __attribute__((ext_vector_type(4)))  unsigned int v4u;

// ---------------------------------------------------------------------------
// Fragment loader: row-major [rows, ld] bf16 source, 16x32 (or 32x16 mirrored)
// fragment per the CDNA5 16-bit A-matrix layout:
//   lane = idx + 16*half; lane holds row `idx`;
//   VGPR0..3 <- K = half*8 + {0..7};  VGPR4..7 <- K = 16 + half*8 + {0..7}
// Both halves are 8 contiguous bf16 => two 16-byte loads.
// ---------------------------------------------------------------------------
__device__ __forceinline__ v16bf load_frag(const bf16_t* p, int row, int ld,
                                           int k0, int half) {
  const bf16_t* r = p + (size_t)row * ld + k0 + half * 8;
  union { v4u u[2]; v16bf v; } u;
  u.u[0] = *(const v4u*)r;        // K rel = half*8 + 0..7
  u.u[1] = *(const v4u*)(r + 16); // K rel = 16 + half*8 + 0..7
  return u.v;
}

// ---------------------------------------------------------------------------
// WMMA GEMM, N fixed at 128, one 16x16 tile per wave, 8 waves/block.
// A: [M,K] bf16 row-major. Bt: [128,K] bf16 (weight pre-transposed).
// ACT: 0=none 1=relu 2=sigmoid 3=tanh
// outF (f32 [M,128]), outB (bf16 [M,128]), outT (f32 [128,ldT] transposed) are
// each optional (nullptr to skip).
// ---------------------------------------------------------------------------
template<int ACT, bool HASBIAS>
__global__ void wmma_gemm_n128(const bf16_t* __restrict__ A,
                               const bf16_t* __restrict__ Bt,
                               const float* __restrict__ bias,
                               float* outF, bf16_t* outB, float* outT, int ldT,
                               int K) {
  const int wave = threadIdx.x >> 5;
  const int lane = threadIdx.x & 31;
  const int idx  = lane & 15;
  const int half = lane >> 4;
  const int mBase = blockIdx.x << 4;
  const int nBase = wave << 4;

  v8f acc = {0.f, 0.f, 0.f, 0.f, 0.f, 0.f, 0.f, 0.f};
  for (int k0 = 0; k0 < K; k0 += 32) {
    v16bf a = load_frag(A,  mBase + idx, K, k0, half);
    v16bf b = load_frag(Bt, nBase + idx, K, k0, half);
    acc = __builtin_amdgcn_wmma_f32_16x16x32_bf16(false, a, false, b,
                                                  (short)0, acc, false, false);
  }

  const int n = nBase + idx;
  const float bv = HASBIAS ? bias[n] : 0.0f;
#pragma unroll
  for (int r = 0; r < 8; ++r) {
    const int m = mBase + r + half * 8;        // C/D layout: VGPR r -> M=r (+8 for hi half)
    float v = acc[r] + bv;
    if (ACT == 1) v = fmaxf(v, 0.0f);
    if (ACT == 2) v = 1.0f / (1.0f + __expf(-v));
    if (ACT == 3) v = tanhf(v);
    if (outF) outF[(size_t)m * CC + n] = v;
    if (outB) outB[(size_t)m * CC + n] = (bf16_t)v;
    if (outT) outT[(size_t)n * ldT + m] = v;
  }
}

// --------------------------- helper kernels --------------------------------

// x [1024,300] f32 -> xb [1024,320] bf16, zero padded K
__global__ void k_xpad(const float* __restrict__ x, bf16_t* __restrict__ xb) {
  int t = blockIdx.x * blockDim.x + threadIdx.x;
  if (t >= NN * FNP) return;
  int r = t / FNP, k = t - r * FNP;
  xb[t] = (k < FN) ? (bf16_t)x[(size_t)r * FN + k] : (bf16_t)0.0f;
}

// W rows [rowoff, rowoff+Ksrc) of [*,128] f32 -> Wt [128, Kpad] bf16 (transposed, padded)
__global__ void k_wt(const float* __restrict__ W, bf16_t* __restrict__ Wt,
                     int Ksrc, int Kpad, int rowoff) {
  int t = blockIdx.x * blockDim.x + threadIdx.x;
  if (t >= CC * Kpad) return;
  int nc = t / Kpad, k = t - nc * Kpad;
  float v = (k < Ksrc) ? W[(size_t)(rowoff + k) * CC + nc] : 0.0f;
  Wt[t] = (bf16_t)v;
}

__global__ void k_deg_init(float* deg) {
  int i = blockIdx.x * blockDim.x + threadIdx.x;
  if (i < NN) deg[i] = 1.0f;                 // self loop
}

__global__ void k_deg_count(const int* __restrict__ dst, float* deg) {
  int e = blockIdx.x * blockDim.x + threadIdx.x;
  if (e < EE) atomicAdd(&deg[dst[e]], 1.0f);
}

__global__ void k_dinv(float* deg) {
  int i = blockIdx.x * blockDim.x + threadIdx.x;
  if (i < NN) deg[i] = rsqrtf(fmaxf(deg[i], 1e-12f));   // in-place -> dinv
}

// agg init with self-loop term: agg[i] = dinv[i]^2 * xw[i]
__global__ void k_selfloop(const float* __restrict__ xwz, const float* __restrict__ xwh,
                           const float* __restrict__ dinv,
                           float* __restrict__ aggz, float* __restrict__ aggh) {
  int t = blockIdx.x * blockDim.x + threadIdx.x;
  if (t >= NN * CC) return;
  int i = t >> 7;
  float s = dinv[i] * dinv[i];
  aggz[t] = s * xwz[t];
  aggh[t] = s * xwh[t];
}

// edge scatter: agg[dst] += dinv[src]*dinv[dst] * xw[src]; 1 thread = 1 edge x 4 ch
__global__ void k_scatter(const int* __restrict__ src, const int* __restrict__ dst,
                          const float* __restrict__ dinv,
                          const float* __restrict__ xwz, const float* __restrict__ xwh,
                          float* __restrict__ aggz, float* __restrict__ aggh) {
  int t = blockIdx.x * blockDim.x + threadIdx.x;
  int e = t >> 5;
  if (e >= EE) return;
  int c4 = (t & 31) << 2;
  int s = src[e], d = dst[e];
  float nm = dinv[s] * dinv[d];
  const float4 vz = *(const float4*)(xwz + (size_t)s * CC + c4);
  const float4 vh = *(const float4*)(xwh + (size_t)s * CC + c4);
  float* pz = aggz + (size_t)d * CC + c4;
  float* ph = aggh + (size_t)d * CC + c4;
  atomicAdd(pz + 0, nm * vz.x); atomicAdd(pz + 1, nm * vz.y);
  atomicAdd(pz + 2, nm * vz.z); atomicAdd(pz + 3, nm * vz.w);
  atomicAdd(ph + 0, nm * vh.x); atomicAdd(ph + 1, nm * vh.y);
  atomicAdd(ph + 2, nm * vh.z); atomicAdd(ph + 3, nm * vh.w);
}

// (agg + gcn bias) -> bf16 for both gates
__global__ void k_agg2bf(const float* __restrict__ aggz, const float* __restrict__ aggh,
                         const float* __restrict__ bz, const float* __restrict__ bh,
                         bf16_t* __restrict__ azb, bf16_t* __restrict__ ahb) {
  int t = blockIdx.x * blockDim.x + threadIdx.x;
  if (t >= NN * CC) return;
  int c = t & (CC - 1);
  azb[t] = (bf16_t)(aggz[t] + bz[c]);
  ahb[t] = (bf16_t)(aggh[t] + bh[c]);
}

// H = (1 - Z) * H_tilde   (H0 = 0)  -> bf16
__global__ void k_hcomb(const float* __restrict__ Z, const float* __restrict__ Ht,
                        bf16_t* __restrict__ Hb) {
  int t = blockIdx.x * blockDim.x + threadIdx.x;
  if (t >= NN * CC) return;
  Hb[t] = (bf16_t)((1.0f - Z[t]) * Ht[t]);
}

// Pair decoder: out[p(i,j)] = sigmoid( sum_c relu(A[i][c]+B[j][c]) * w2[c] + b2 )
// A: [1024,128] (bias folded in), Bt: [128,1024] transposed for coalesced j.
__global__ void k_pairs(const float* __restrict__ A, const float* __restrict__ Bt,
                        const float* __restrict__ w2, const float* __restrict__ b2,
                        float* __restrict__ out) {
  const int i = blockIdx.x;
  __shared__ float As[CC];
  __shared__ float Ws[CC];
  for (int c = threadIdx.x; c < CC; c += blockDim.x) {
    As[c] = A[(size_t)i * CC + c];
    Ws[c] = w2[c];
  }
  __syncthreads();
  if (i >= NN - 1) return;
  const float bb = b2[0];
  const long long base = (long long)i * (NN - 1) - (long long)i * (i - 1) / 2;
  for (int j = i + 1 + threadIdx.x; j < NN; j += blockDim.x) {
    float acc = bb;
#pragma unroll 16
    for (int c = 0; c < CC; ++c)
      acc += fmaxf(As[c] + Bt[(size_t)c * NN + j], 0.0f) * Ws[c];
    out[base + (j - i - 1)] = 1.0f / (1.0f + __expf(-acc));
  }
}

// ---------------------------------------------------------------------------

extern "C" void kernel_launch(void* const* d_in, const int* in_sizes, int n_in,
                              void* d_out, int out_size, void* d_ws, size_t ws_size,
                              hipStream_t stream) {
  const float* x      = (const float*)d_in[0];
  const int*   eidx   = (const int*)  d_in[1];   // [2,E]: src=eidx, dst=eidx+EE
  // d_in[2] edge_attr, d_in[5/6] edge_W/b: dead (edge encoder unused downstream)
  const float* node_W = (const float*)d_in[3];
  const float* node_b = (const float*)d_in[4];
  const float* Wz     = (const float*)d_in[7];
  const float* bz     = (const float*)d_in[8];
  // d_in[9/10] Wr/br, d_in[15/16] Lr_W/b: dead (H0==0 => H*R==0, R unused)
  const float* Wh     = (const float*)d_in[11];
  const float* bh     = (const float*)d_in[12];
  const float* LzW    = (const float*)d_in[13];
  const float* Lzb    = (const float*)d_in[14];
  const float* LhW    = (const float*)d_in[17];
  const float* Lhb    = (const float*)d_in[18];
  const float* d1W    = (const float*)d_in[19];
  const float* d1b    = (const float*)d_in[20];
  const float* d2W    = (const float*)d_in[21];
  const float* d2b    = (const float*)d_in[22];
  float* out = (float*)d_out;

  // -------- workspace carve-out (256B aligned) --------
  char* w = (char*)d_ws;
  size_t off = 0;
  auto take = [&](size_t bytes) -> void* {
    void* p = w + off;
    off = (off + bytes + 255) & ~(size_t)255;
    return p;
  };
  bf16_t* xb    = (bf16_t*)take((size_t)NN * FNP * 2);
  bf16_t* nWt   = (bf16_t*)take((size_t)CC * FNP * 2);
  bf16_t* xencb = (bf16_t*)take((size_t)NN * CC * 2);
  bf16_t* Wzt   = (bf16_t*)take((size_t)CC * CC * 2);
  bf16_t* Wht   = (bf16_t*)take((size_t)CC * CC * 2);
  bf16_t* Lzt   = (bf16_t*)take((size_t)CC * CC * 2);
  bf16_t* Lht   = (bf16_t*)take((size_t)CC * CC * 2);
  bf16_t* d1at  = (bf16_t*)take((size_t)CC * CC * 2);
  bf16_t* d1bt  = (bf16_t*)take((size_t)CC * CC * 2);
  float*  xwz   = (float*) take((size_t)NN * CC * 4);
  float*  xwh   = (float*) take((size_t)NN * CC * 4);
  float*  dinv  = (float*) take((size_t)NN * 4);
  float*  aggz  = (float*) take((size_t)NN * CC * 4);
  float*  aggh  = (float*) take((size_t)NN * CC * 4);
  bf16_t* azb   = (bf16_t*)take((size_t)NN * CC * 2);
  bf16_t* ahb   = (bf16_t*)take((size_t)NN * CC * 2);
  float*  Zf    = (float*) take((size_t)NN * CC * 4);
  float*  Htf   = (float*) take((size_t)NN * CC * 4);
  bf16_t* Hb    = (bf16_t*)take((size_t)NN * CC * 2);
  float*  Adec  = (float*) take((size_t)NN * CC * 4);
  float*  BtD   = (float*) take((size_t)CC * NN * 4);

  const int T = 256;
  const int* srcI = eidx;
  const int* dstI = eidx + EE;

  // -------- stage 0: pack/transpose weights & inputs to bf16 --------
  k_xpad<<<(NN * FNP + T - 1) / T, T, 0, stream>>>(x, xb);
  k_wt<<<(CC * FNP + T - 1) / T, T, 0, stream>>>(node_W, nWt, FN, FNP, 0);
  k_wt<<<(CC * CC + T - 1) / T, T, 0, stream>>>(Wz,  Wzt,  CC, CC, 0);
  k_wt<<<(CC * CC + T - 1) / T, T, 0, stream>>>(Wh,  Wht,  CC, CC, 0);
  k_wt<<<(CC * CC + T - 1) / T, T, 0, stream>>>(LzW, Lzt,  CC, CC, 0);    // rows 0..127 only (H==0)
  k_wt<<<(CC * CC + T - 1) / T, T, 0, stream>>>(LhW, Lht,  CC, CC, 0);
  k_wt<<<(CC * CC + T - 1) / T, T, 0, stream>>>(d1W, d1at, CC, CC, 0);    // dec1_W[:128]
  k_wt<<<(CC * CC + T - 1) / T, T, 0, stream>>>(d1W, d1bt, CC, CC, CC);   // dec1_W[128:]

  // -------- stage 1: node encoder  x_enc = relu(x @ node_W + b) --------
  wmma_gemm_n128<1, true><<<NN / 16, T, 0, stream>>>(xb, nWt, node_b,
                                                     nullptr, xencb, nullptr, 0, FNP);

  // -------- stage 2: GCN xW GEMMs --------
  wmma_gemm_n128<0, false><<<NN / 16, T, 0, stream>>>(xencb, Wzt, nullptr,
                                                      xwz, nullptr, nullptr, 0, CC);
  wmma_gemm_n128<0, false><<<NN / 16, T, 0, stream>>>(xencb, Wht, nullptr,
                                                      xwh, nullptr, nullptr, 0, CC);

  // -------- stage 3: degrees + symmetric-norm scatter aggregation --------
  k_deg_init<<<(NN + T - 1) / T, T, 0, stream>>>(aggz /*unused*/ == nullptr ? nullptr : dinv); // deg in dinv buf
  k_deg_count<<<(EE + T - 1) / T, T, 0, stream>>>(dstI, dinv);
  k_dinv<<<(NN + T - 1) / T, T, 0, stream>>>(dinv);
  k_selfloop<<<(NN * CC + T - 1) / T, T, 0, stream>>>(xwz, xwh, dinv, aggz, aggh);
  k_scatter<<<(EE * 32 + T - 1) / T, T, 0, stream>>>(srcI, dstI, dinv, xwz, xwh, aggz, aggh);
  k_agg2bf<<<(NN * CC + T - 1) / T, T, 0, stream>>>(aggz, aggh, bz, bh, azb, ahb);

  // -------- stage 4: gates  Z = sigmoid(.. @ Lz), H~ = tanh(.. @ Lh) --------
  wmma_gemm_n128<2, true><<<NN / 16, T, 0, stream>>>(azb, Lzt, Lzb,
                                                     Zf, nullptr, nullptr, 0, CC);
  wmma_gemm_n128<3, true><<<NN / 16, T, 0, stream>>>(ahb, Lht, Lhb,
                                                     Htf, nullptr, nullptr, 0, CC);
  k_hcomb<<<(NN * CC + T - 1) / T, T, 0, stream>>>(Zf, Htf, Hb);

  // -------- stage 5: decoder factorization  A = H@W1a + b1,  Bt = (H@W1b)^T --------
  wmma_gemm_n128<0, true><<<NN / 16, T, 0, stream>>>(Hb, d1at, d1b,
                                                     Adec, nullptr, nullptr, 0, CC);
  wmma_gemm_n128<0, false><<<NN / 16, T, 0, stream>>>(Hb, d1bt, nullptr,
                                                      nullptr, nullptr, BtD, NN, CC);

  // -------- stage 6: all-pairs epilogue --------
  k_pairs<<<NN, T, 0, stream>>>(Adec, BtD, d2W, d2b, out);
}
